// MyGENConv_14259291423280
// MI455X (gfx1250) — matrix operand order repaired and checked by reading, not compiled, not run on hardware
//
#include <hip/hip_runtime.h>

// CDNA5 / gfx1250 wave32 WMMA types
typedef __attribute__((ext_vector_type(16))) _Float16 v16h;
typedef __attribute__((ext_vector_type(8)))  float    v8f;

__device__ __forceinline__ v8f wmma_f16(v16h a, v16h b, v8f c) {
    // D = A(16x32 f16) * B(32x16 f16) + C(16x16 f32)
    return __builtin_amdgcn_wmma_f32_16x16x32_f16(
        /*neg_a=*/false, a, /*neg_b=*/false, b,
        /*c_mod=*/(short)0, c, /*reuse_a=*/false, /*reuse_b=*/false);
}

// Build one A fragment (16x32 slice starting at kBase) from a row-major f32
// matrix. ISA 16-bit A layout: lanes 0-15 -> row=lane, halves 0..7 = K+0..7,
// halves 8..15 = K+16..23 ; lanes 16-31 -> +8 on the K sub-offset.
// Offsets kept in u32 so ISel uses SGPR-base + u32-voffset (scale_offset).
__device__ __forceinline__ v16h load_a_f32(const float* __restrict__ base,
                                           unsigned ld, unsigned kBase,
                                           unsigned lane) {
    unsigned row = lane & 15u;
    unsigned grp = lane >> 4;
    unsigned off = row * ld + kBase + grp * 8u;
    v16h a;
#pragma unroll
    for (int i = 0; i < 8; ++i) a[i] = (_Float16)base[off + i];
#pragma unroll
    for (int i = 0; i < 8; ++i) a[i + 8] = (_Float16)base[off + 16 + i];
    return a;
}

// Pre-swizzle a KxNw row-major f32 weight matrix into per-lane WMMA B
// fragments in LDS (f16). Fragment (t, ks): columns 16t..16t+15, K 32ks..+31.
// ISA B layout: lanes 0-15 hold K 0..7,16..23, lanes 16-31 hold K 8..15,
// 24..31, lane selects column. Each lane-slot is 32B contiguous.
__device__ __forceinline__ void stage_b(const float* __restrict__ W,
                                        unsigned Nw, unsigned tiles,
                                        unsigned ksteps, v16h* bLds) {
    unsigned total = tiles * ksteps * 32u;
    for (unsigned i = threadIdx.x; i < total; i += blockDim.x) {
        unsigned s  = i & 31u;
        unsigned ks = (i >> 5) % ksteps;
        unsigned t  = i / (32u * ksteps);
        unsigned col  = t * 16u + (s & 15u);
        unsigned kOff = ks * 32u + (s >> 4) * 8u;
        v16h b;
#pragma unroll
        for (int h = 0; h < 16; ++h)
            b[h] = (_Float16)W[(kOff + h + (h < 8 ? 0 : 8)) * Nw + col];
        bLds[i] = b;
    }
}

// ---------------------------------------------------------------------------
// K0: zero-fill
__global__ void k_zero(float* __restrict__ p, unsigned n) {
    for (unsigned i = blockIdx.x * blockDim.x + threadIdx.x; i < n;
         i += gridDim.x * blockDim.x)
        p[i] = 0.0f;
}

// ---------------------------------------------------------------------------
// K1: fused edge pass.
// msg = relu(x[src] + edge_attr @ W_edge) + 1e-7  (16-edge tile per wave, WMMA)
// then atomically accumulate denom += exp(msg), numer += msg*exp(msg) at dst.
// (max-shift dropped: msg in (0, ~10], exp-safe; result mathematically equal)
__global__ __launch_bounds__(256) void k_edge(
    const float* __restrict__ x, const int* __restrict__ src,
    const int* __restrict__ dst, const float* __restrict__ edge_attr,
    const float* __restrict__ W_edge, float* __restrict__ numer,
    float* __restrict__ denom, int E) {
    __shared__ v16h bLds[8 * 2 * 32];  // 16 KB: W_edge (64x128) fragments
    stage_b(W_edge, 128u, /*tiles=*/8u, /*ksteps=*/2u, bLds);
    __syncthreads();

    const unsigned lane = threadIdx.x & 31u;
    const unsigned wave = threadIdx.x >> 5;
    const int nTiles = E >> 4;
    const int step = gridDim.x * 8;

    for (int tile = (int)(blockIdx.x * 8 + wave); tile < nTiles; tile += step) {
        const unsigned eBase = (unsigned)tile << 4;
        const float* aBase = edge_attr + (size_t)eBase * 64u;
        // prefetch next tile's edge_attr (speculative, global_prefetch_b8)
        __builtin_prefetch(aBase + (size_t)step * 16 * 64, 0, 1);

        v16h a0 = load_a_f32(aBase, 64u, 0u, lane);
        v16h a1 = load_a_f32(aBase, 64u, 32u, lane);

        // row-base offsets: invariant across the 8 N-tiles
        const unsigned rowG = (lane >> 4) * 8u;
        unsigned srcOff[8], dstOff[8];
#pragma unroll
        for (int r = 0; r < 8; ++r) {
            unsigned e = eBase + rowG + (unsigned)r;
            srcOff[r] = (unsigned)src[e] * 128u;
            dstOff[r] = (unsigned)dst[e] * 128u;
        }

#pragma unroll
        for (int t = 0; t < 8; ++t) {
            v8f c = {};
            c = wmma_f16(a0, bLds[(t * 2 + 0) * 32 + lane], c);
            c = wmma_f16(a1, bLds[(t * 2 + 1) * 32 + lane], c);
            const unsigned col = (unsigned)t * 16u + (lane & 15u);

            // 1) batch the 8 residual gathers (one load clause, one wait)
            float xv[8];
#pragma unroll
            for (int r = 0; r < 8; ++r) xv[r] = x[srcOff[r] + col];

            // 2) compute all exp chains (pipelines on trans unit)
            float ev[8], mv[8];
#pragma unroll
            for (int r = 0; r < 8; ++r) {
                float v = c[r] + xv[r];
                v = fmaxf(v, 0.0f) + 1e-7f;
                ev[r] = __expf(v);
                mv[r] = v * ev[r];
            }

            // 3) issue all 16 atomics back-to-back
#pragma unroll
            for (int r = 0; r < 8; ++r) {
                unsigned o = dstOff[r] + col;
                atomicAdd(denom + o, ev[r]);
                atomicAdd(numer + o, mv[r]);
            }
        }
    }
}

// ---------------------------------------------------------------------------
// K2: out = numer/(denom+1e-16) + x ; h = out @ W1  (128 -> 256), WMMA
__global__ __launch_bounds__(256) void k_mlp1(
    const float* __restrict__ x, const float* __restrict__ numer,
    const float* __restrict__ denom, const float* __restrict__ W1,
    float* __restrict__ h, int N) {
    __shared__ v16h bLds[16 * 4 * 32];  // 64 KB: W1 (128x256) fragments
    stage_b(W1, 256u, /*tiles=*/16u, /*ksteps=*/4u, bLds);
    __syncthreads();

    const unsigned lane = threadIdx.x & 31u;
    const unsigned wave = threadIdx.x >> 5;
    const int nTiles = N >> 4;

    for (int tile = (int)(blockIdx.x * 8 + wave); tile < nTiles;
         tile += gridDim.x * 8) {
        const unsigned nBase = (unsigned)tile << 4;
        const unsigned rowOff = (nBase + (lane & 15u)) * 128u;
        const unsigned grp = lane >> 4;

        v16h a[4];
#pragma unroll
        for (int ks = 0; ks < 4; ++ks) {
#pragma unroll
            for (int i = 0; i < 16; ++i) {
                unsigned k = (unsigned)ks * 32u + grp * 8u +
                             (unsigned)((i < 8) ? i : (i + 8));
                unsigned o = rowOff + k;
                float v = numer[o] / (denom[o] + 1e-16f) + x[o];
                a[ks][i] = (_Float16)v;
            }
        }

        // output row offsets: invariant across the 16 N-tiles
        unsigned outOff[8];
#pragma unroll
        for (int r = 0; r < 8; ++r)
            outOff[r] = (nBase + (unsigned)r + grp * 8u) * 256u;

#pragma unroll
        for (int t = 0; t < 16; ++t) {
            v8f c = {};
#pragma unroll
            for (int ks = 0; ks < 4; ++ks)
                c = wmma_f16(a[ks], bLds[(t * 4 + ks) * 32 + lane], c);
            const unsigned col = (unsigned)t * 16u + (lane & 15u);
#pragma unroll
            for (int r = 0; r < 8; ++r)
                h[outOff[r] + col] = c[r];
        }
    }
}

// ---------------------------------------------------------------------------
// K3: per-channel batch sums (thread = channel, 256 channels)
__global__ __launch_bounds__(256) void k_bnstats(const float* __restrict__ h,
                                                 float* __restrict__ sumA,
                                                 float* __restrict__ sqA,
                                                 int N) {
    const unsigned c = threadIdx.x;
    float s = 0.0f, q = 0.0f;
    for (unsigned r = blockIdx.x; r < (unsigned)N; r += gridDim.x) {
        float v = h[r * 256u + c];
        s += v;
        q += v * v;
    }
    atomicAdd(sumA + c, s);
    atomicAdd(sqA + c, q);
}

// K4: in-place BN (training stats, biased var) + ReLU:  h = relu(h*s + b)
__global__ __launch_bounds__(256) void k_bnnorm(
    float* __restrict__ h, const float* __restrict__ sumA,
    const float* __restrict__ sqA, const float* __restrict__ gamma,
    const float* __restrict__ beta, int N) {
    const unsigned c = threadIdx.x;
    const float inv = 1.0f / (float)N;
    const float m = sumA[c] * inv;
    const float var = sqA[c] * inv - m * m;
    const float s = gamma[c] * rsqrtf(var + 1e-5f);
    const float b = beta[c] - m * s;
    for (unsigned r = blockIdx.x; r < (unsigned)N; r += gridDim.x) {
        unsigned o = r * 256u + c;
        h[o] = fmaxf(h[o] * s + b, 0.0f);
    }
}

// ---------------------------------------------------------------------------
// K5: d_out = h_norm @ W2  (256 -> 128), WMMA
__global__ __launch_bounds__(256) void k_mlp2(const float* __restrict__ h,
                                              const float* __restrict__ W2,
                                              float* __restrict__ out, int N) {
    __shared__ v16h bLds[8 * 8 * 32];  // 64 KB: W2 (256x128) fragments
    stage_b(W2, 128u, /*tiles=*/8u, /*ksteps=*/8u, bLds);
    __syncthreads();

    const unsigned lane = threadIdx.x & 31u;
    const unsigned wave = threadIdx.x >> 5;
    const int nTiles = N >> 4;

    for (int tile = (int)(blockIdx.x * 8 + wave); tile < nTiles;
         tile += gridDim.x * 8) {
        const unsigned nBase = (unsigned)tile << 4;
        const float* aBase = h + (size_t)nBase * 256u;
        const unsigned grp = lane >> 4;

        v16h a[8];
#pragma unroll
        for (int ks = 0; ks < 8; ++ks)
            a[ks] = load_a_f32(aBase, 256u, (unsigned)ks * 32u, lane);

        unsigned outOff[8];
#pragma unroll
        for (int r = 0; r < 8; ++r)
            outOff[r] = (nBase + (unsigned)r + grp * 8u) * 128u;

#pragma unroll
        for (int t = 0; t < 8; ++t) {
            v8f c = {};
#pragma unroll
            for (int ks = 0; ks < 8; ++ks)
                c = wmma_f16(a[ks], bLds[(t * 8 + ks) * 32 + lane], c);
            const unsigned col = (unsigned)t * 16u + (lane & 15u);
#pragma unroll
            for (int r = 0; r < 8; ++r)
                out[outOff[r] + col] = c[r];
        }
    }
}

// ---------------------------------------------------------------------------
extern "C" void kernel_launch(void* const* d_in, const int* in_sizes, int n_in,
                              void* d_out, int out_size, void* d_ws,
                              size_t ws_size, hipStream_t stream) {
    const float* x      = (const float*)d_in[0];  // [N,128]
    const int*   ei     = (const int*)d_in[1];    // [2,E]
    const float* ea     = (const float*)d_in[2];  // [E,64]
    const float* W_edge = (const float*)d_in[3];  // [64,128]
    const float* W1     = (const float*)d_in[4];  // [128,256]
    const float* gamma  = (const float*)d_in[5];  // [256]
    const float* beta   = (const float*)d_in[6];  // [256]
    const float* W2     = (const float*)d_in[7];  // [256,128]
    float* out = (float*)d_out;

    const int N = in_sizes[0] / 128;
    const int E = in_sizes[1] / 2;
    const int* src = ei;
    const int* dst = ei + E;

    // workspace layout (floats): numer[N*128] | denom[N*128] | h[N*256] | sum[256] | sq[256]
    float* ws    = (float*)d_ws;
    float* numer = ws;
    float* denom = ws + (size_t)N * 128;
    float* h     = ws + (size_t)2 * N * 128;
    float* sumA  = h + (size_t)N * 256;
    float* sqA   = sumA + 256;

    k_zero<<<512, 256, 0, stream>>>(numer, (unsigned)(2 * N * 128));
    k_zero<<<1, 256, 0, stream>>>(sumA, 512u);
    k_edge<<<1024, 256, 0, stream>>>(x, src, dst, ea, W_edge, numer, denom, E);
    k_mlp1<<<160, 256, 0, stream>>>(x, numer, denom, W1, h, N);
    k_bnstats<<<256, 256, 0, stream>>>(h, sumA, sqA, N);
    k_bnnorm<<<256, 256, 0, stream>>>(h, sumA, sqA, gamma, beta, N);
    k_mlp2<<<160, 256, 0, stream>>>(h, W2, out, N);
}
